// Transformer_50465865728507
// MI455X (gfx1250) — matrix operand order repaired and checked by reading, not compiled, not run on hardware
//
#include <hip/hip_runtime.h>
#include <hip/hip_bf16.h>
#include <math.h>

// ---------------- problem constants ----------------
#define BB      2
#define NSEQ    2048
#define DIMM    1024
#define HH      16
#define DHH     64
#define NNUL    2
#define JJ      (NSEQ + NNUL)   // 2050
#define JP      2080            // padded J (multiple of 32; 2080*2B row stride is 16B-aligned)
#define INNER   (HH * DHH)      // 1024
#define SCALE_F 8.0f
#define LN_EPS_F 1e-5f

typedef __bf16 bf16;
typedef __attribute__((ext_vector_type(16))) __bf16 v16bf;
typedef __attribute__((ext_vector_type(8)))  __bf16 v8bf;
typedef __attribute__((ext_vector_type(8)))  float  v8f;

// ---------------- small helpers ----------------
__device__ __forceinline__ bf16 f2bf(float f) {
  unsigned u = __builtin_bit_cast(unsigned, f);
  u += 0x7FFFu + ((u >> 16) & 1u);              // round-to-nearest-even
  unsigned short s = (unsigned short)(u >> 16);
  return __builtin_bit_cast(bf16, s);
}

__device__ __forceinline__ v8f vzero8() {
  v8f z;
#pragma unroll
  for (int i = 0; i < 8; ++i) z[i] = 0.0f;
  return z;
}

// A-matrix fragment (16x32 bf16, row-major source, leading dim `ld` elements).
// ISA 7.12.2: lane L (m=L%16, hi=L/16) holds K = hi*8+{0..7} and 16+hi*8+{0..7}.
__device__ __forceinline__ v16bf load_a_frag(const bf16* __restrict__ base, int ld) {
  const int lane = threadIdx.x & 31;
  const int row  = lane & 15;
  const int hi   = lane >> 4;
  const bf16* p = base + (size_t)row * ld + hi * 8;
  v8bf lo = *reinterpret_cast<const v8bf*>(p);
  v8bf hv = *reinterpret_cast<const v8bf*>(p + 16);
  v16bf r;
#pragma unroll
  for (int e = 0; e < 8; ++e) { r[e] = lo[e]; r[8 + e] = hv[e]; }
  return r;
}

// B-matrix fragment (32x16 bf16). `base` points at B^T tile origin, i.e. each
// B column (16 elems of N) is a contiguous row of length-ld in memory.
// Same K order as the A fragment: lane L (n=L%16, hi=L/16)
// holds K = hi*8+{0..7} and 16+hi*8+{0..7}.
__device__ __forceinline__ v16bf load_b_frag(const bf16* __restrict__ base, int ld) {
  const int lane = threadIdx.x & 31;
  const int col  = lane & 15;
  const int hi   = lane >> 4;
  const bf16* p = base + (size_t)col * ld + hi * 8;
  v8bf lo = *reinterpret_cast<const v8bf*>(p);
  v8bf hv = *reinterpret_cast<const v8bf*>(p + 16);
  v16bf r;
#pragma unroll
  for (int e = 0; e < 8; ++e) { r[e] = lo[e]; r[8 + e] = hv[e]; }
  return r;
}

__device__ __forceinline__ v8f wmma_bf16(v16bf a, v16bf b, v8f c) {
  return __builtin_amdgcn_wmma_f32_16x16x32_bf16(false, a, false, b, (short)0, c,
                                                 false, false);
}

// ---------------- kernel 1: LayerNorm + bf16 casts ----------------
__global__ __launch_bounds__(256) void ln_prep_kernel(
    const float* __restrict__ x, const float* __restrict__ g,
    const float* __restrict__ bta, bf16* __restrict__ xb, bf16* __restrict__ xnb) {
  const int row = blockIdx.x;                 // b*N + n
  const int t   = threadIdx.x;
  const float* xr = x + (size_t)row * DIMM;
  float4 v = reinterpret_cast<const float4*>(xr)[t];
  float s  = v.x + v.y + v.z + v.w;
  float ss = v.x * v.x + v.y * v.y + v.z * v.z + v.w * v.w;
  __shared__ float r1[256], r2[256];
  r1[t] = s; r2[t] = ss;
  __syncthreads();
  for (int off = 128; off > 0; off >>= 1) {
    if (t < off) { r1[t] += r1[t + off]; r2[t] += r2[t + off]; }
    __syncthreads();
  }
  const float mu  = r1[0] * (1.0f / DIMM);
  const float var = r2[0] * (1.0f / DIMM) - mu * mu;
  const float rs  = rsqrtf(var + LN_EPS_F);
  const int c = t * 4;
  float xs[4] = {v.x, v.y, v.z, v.w};
#pragma unroll
  for (int e = 0; e < 4; ++e) {
    float xn = (xs[e] - mu) * rs * g[c + e] + bta[c + e];
    xnb[(size_t)row * DIMM + c + e] = f2bf(xn);
    xb [(size_t)row * DIMM + c + e] = f2bf(xs[e]);
  }
}

// ---------------- kernel 2: weight cast + transpose (W[K][N] -> Wt[N][K] bf16)
__global__ __launch_bounds__(256) void cvt_transpose_kernel(
    const float* __restrict__ W, bf16* __restrict__ Wt, int K, int Nc) {
  const long total = (long)K * Nc;
  for (long idx = (long)blockIdx.x * blockDim.x + threadIdx.x; idx < total;
       idx += (long)gridDim.x * blockDim.x) {
    long k = idx / Nc, n = idx % Nc;
    Wt[n * (long)K + k] = f2bf(W[idx]);
  }
}

// ---------------- kernel 3: bf16 WMMA GEMM: C[M][Nc] = A[M][K] * Bt[Nc][K]^T
// The 32-row A panel is shared by all 8 waves of the block: stage it through
// LDS with double-buffered GLOBAL_LOAD_ASYNC_TO_LDS_B128 (ASYNCcnt-tracked),
// cutting A global traffic 8x. B fragments stay direct global_load_b128.
__global__ __launch_bounds__(256) void gemm_bf16_kernel(
    const bf16* __restrict__ A, const bf16* __restrict__ Bt,
    float* __restrict__ C, int M, int Nc, int K) {
  const int t    = threadIdx.x;
  const int wave = t >> 5;
  const int lane = t & 31;
  const int col  = lane & 15;
  const int hi   = lane >> 4;
  const int m0 = blockIdx.y * 32;
  const int n0 = blockIdx.x * 256 + wave * 32;

  __shared__ __align__(16) bf16 abuf[2][32 * 64];   // 2 x 4KB A tiles (32 rows x 64 K)

  // each of 256 threads copies one 16B chunk of the 32x64 tile
  const int arow = t >> 3;          // 0..31
  const int acol = (t & 7) * 8;     // element offset, 8 bf16 = 16B

  auto stage = [&](int bi, int kc) {
    const bf16* gp = A + (size_t)(m0 + arow) * K + kc + acol;
    unsigned ldsoff = (unsigned)(uintptr_t)(&abuf[bi][arow * 64 + acol]);
    asm volatile("global_load_async_to_lds_b128 %0, %1, off"
                 :: "v"(ldsoff), "v"((unsigned long long)(uintptr_t)gp)
                 : "memory");
  };

  v8f c00 = vzero8(), c01 = vzero8(), c10 = vzero8(), c11 = vzero8();

  const int CHUNKS = K / 64;
  stage(0, 0);
  for (int c = 0; c < CHUNKS; ++c) {
    const int kc = c * 64;
    if (c + 1 < CHUNKS) {
      stage((c + 1) & 1, kc + 64);
      asm volatile("s_wait_asynccnt 0x1" ::: "memory");  // chunk c landed (in-order)
    } else {
      asm volatile("s_wait_asynccnt 0x0" ::: "memory");
    }
    __syncthreads();                                     // visible to all waves
    const bf16* as = &abuf[c & 1][0];
#pragma unroll
    for (int kk = 0; kk < 64; kk += 32) {
      v16bf a0 = load_a_frag(as + kk, 64);               // rows m0..m0+15
      v16bf a1 = load_a_frag(as + 16 * 64 + kk, 64);     // rows m0+16..m0+31
      v16bf b0 = load_b_frag(Bt + (size_t)n0 * K + kc + kk, K);
      v16bf b1 = load_b_frag(Bt + (size_t)(n0 + 16) * K + kc + kk, K);
      c00 = wmma_bf16(a0, b0, c00);
      c01 = wmma_bf16(a0, b1, c01);
      c10 = wmma_bf16(a1, b0, c10);
      c11 = wmma_bf16(a1, b1, c11);
    }
    __syncthreads();                                     // done reading before overwrite
  }
#pragma unroll
  for (int r = 0; r < 8; ++r) {
    const int mr = m0 + r + 8 * hi;
    C[(size_t)mr * Nc + n0 + col]             = c00[r];
    C[(size_t)mr * Nc + n0 + 16 + col]        = c01[r];
    C[(size_t)(mr + 16) * Nc + n0 + col]      = c10[r];
    C[(size_t)(mr + 16) * Nc + n0 + 16 + col] = c11[r];
  }
}

// ---------------- kernel 4a: Q l2norm + scale -> qb[b][h][n][dh] bf16
__global__ __launch_bounds__(256) void norm_q_kernel(
    const float* __restrict__ qf, const float* __restrict__ qscale,
    bf16* __restrict__ qb) {
  const int lane = threadIdx.x & 31;
  const long row = (long)blockIdx.x * 8 + (threadIdx.x >> 5);  // (b, n, h)
  if (row >= (long)BB * NSEQ * HH) return;
  const int  h  = (int)(row % HH);
  const long bn = row / HH;
  const float* src = qf + bn * INNER + h * DHH;
  float a0 = src[lane], a1 = src[lane + 32];
  float ss = a0 * a0 + a1 * a1;
#pragma unroll
  for (int m = 16; m >= 1; m >>= 1) ss += __shfl_xor(ss, m, 32);
  const float inv = 1.0f / fmaxf(sqrtf(ss), 1e-12f);
  const long b = bn / NSEQ, n = bn % NSEQ;
  bf16* dst = qb + (((long)b * HH + h) * NSEQ + n) * DHH;
  dst[lane]      = f2bf(a0 * inv * qscale[lane]);
  dst[lane + 32] = f2bf(a1 * inv * qscale[lane + 32]);
}

// ---------------- kernel 4b: K (null + projected) l2norm + scale -> kb[b][h][JP][dh]
__global__ __launch_bounds__(256) void norm_k_kernel(
    const float* __restrict__ kvf, const float* __restrict__ nullkv,
    const float* __restrict__ kscale, bf16* __restrict__ kb) {
  const int lane = threadIdx.x & 31;
  const long row = (long)blockIdx.x * 8 + (threadIdx.x >> 5);  // (b, h, jj)
  if (row >= (long)BB * HH * JJ) return;
  const long b  = row / ((long)HH * JJ);
  const long rm = row % ((long)HH * JJ);
  const int  h  = (int)(rm / JJ);
  const int  jj = (int)(rm % JJ);
  float a0, a1;
  if (jj < NNUL) {  // null_kv reshaped (H, NNUL, 2, DH); slot 0 = K
    const float* s = nullkv + (((long)h * NNUL + jj) * 2 + 0) * DHH;
    a0 = s[lane]; a1 = s[lane + 32];
  } else {
    const long n = jj - NNUL;
    const float* s = kvf + ((long)b * NSEQ + n) * (2 * INNER) + h * DHH;
    a0 = s[lane]; a1 = s[lane + 32];
  }
  float ss = a0 * a0 + a1 * a1;
#pragma unroll
  for (int m = 16; m >= 1; m >>= 1) ss += __shfl_xor(ss, m, 32);
  const float inv = 1.0f / fmaxf(sqrtf(ss), 1e-12f);
  bf16* dst = kb + (((long)b * HH + h) * JP + jj) * DHH;
  dst[lane]      = f2bf(a0 * inv * kscale[lane]);
  dst[lane + 32] = f2bf(a1 * inv * kscale[lane + 32]);
}

// ---------------- kernel 4c: V transposed -> vt[b][h][dh][JP] bf16
__global__ __launch_bounds__(256) void norm_v_kernel(
    const float* __restrict__ kvf, const float* __restrict__ nullkv,
    bf16* __restrict__ vt) {
  const int lane = threadIdx.x & 31;
  const long row = (long)blockIdx.x * 8 + (threadIdx.x >> 5);  // (b, h, jj)
  if (row >= (long)BB * HH * JJ) return;
  const long b  = row / ((long)HH * JJ);
  const long rm = row % ((long)HH * JJ);
  const int  h  = (int)(rm / JJ);
  const int  jj = (int)(rm % JJ);
  float a0, a1;
  if (jj < NNUL) {  // slot 1 = V
    const float* s = nullkv + (((long)h * NNUL + jj) * 2 + 1) * DHH;
    a0 = s[lane]; a1 = s[lane + 32];
  } else {
    const long n = jj - NNUL;
    const float* s = kvf + ((long)b * NSEQ + n) * (2 * INNER) + INNER + h * DHH;
    a0 = s[lane]; a1 = s[lane + 32];
  }
  bf16* base = vt + (((long)b * HH + h) * DHH) * JP;
  base[(long)(lane)      * JP + jj] = f2bf(a0);
  base[(long)(lane + 32) * JP + jj] = f2bf(a1);
}

// zero the vt padding columns jj in [JJ, JP) so masked P (==0) never meets NaN
__global__ __launch_bounds__(256) void pad_v_kernel(bf16* __restrict__ vt) {
  const long total = (long)BB * HH * DHH * (JP - JJ);
  for (long idx = (long)blockIdx.x * blockDim.x + threadIdx.x; idx < total;
       idx += (long)gridDim.x * blockDim.x) {
    const long r  = idx / (JP - JJ);
    const int  jj = JJ + (int)(idx % (JP - JJ));
    vt[r * JP + jj] = f2bf(0.0f);
  }
}

// ---------------- kernel 5: flash attention (bf16 WMMA, online softmax)
__global__ __launch_bounds__(128) void attn_kernel(
    const bf16* __restrict__ qb, const bf16* __restrict__ kb,
    const bf16* __restrict__ vt, bf16* __restrict__ aob) {
  const int b    = blockIdx.z;
  const int h    = blockIdx.y;
  const int wave = threadIdx.x >> 5;
  const int lane = threadIdx.x & 31;
  const int col  = lane & 15;
  const int hi   = lane >> 4;
  const int m0   = blockIdx.x * 64 + wave * 16;   // first query row of this wave

  const bf16* qrow = qb + (((size_t)b * HH + h) * NSEQ) * DHH;
  const bf16* krow = kb + (((size_t)b * HH + h) * JP) * DHH;
  const bf16* vrow = vt + (((size_t)b * HH + h) * DHH) * JP;

  const float slope = exp2f(-0.5f * (float)(h + 1));  // alibi slope for 16 heads

  const v16bf qa0 = load_a_frag(qrow + (size_t)m0 * DHH + 0,  DHH);
  const v16bf qa1 = load_a_frag(qrow + (size_t)m0 * DHH + 32, DHH);

  v8f o0 = vzero8(), o1 = vzero8(), o2 = vzero8(), o3 = vzero8();
  float rmax[8], rsum[8];
#pragma unroll
  for (int r = 0; r < 8; ++r) { rmax[r] = -INFINITY; rsum[r] = 0.0f; }

  __shared__ __align__(16) bf16 pshm[4][16 * 32];   // per-wave P tile (16x32)
  bf16* pw = &pshm[wave][0];

  const int jlast   = m0 + 15 + NNUL;       // last visible kv column for this tile
  const int nblocks = jlast / 32 + 1;

  for (int cb = 0; cb < nblocks; ++cb) {
    const int jj0 = cb * 32;
    // ---- S = q * k^T for two 16-col tiles (K = dh = 64 -> 2 WMMAs each)
    v8f s0 = vzero8(), s1 = vzero8();
    s0 = wmma_bf16(qa0, load_b_frag(krow + (size_t)jj0 * DHH + 0,  DHH), s0);
    s0 = wmma_bf16(qa1, load_b_frag(krow + (size_t)jj0 * DHH + 32, DHH), s0);
    s1 = wmma_bf16(qa0, load_b_frag(krow + (size_t)(jj0 + 16) * DHH + 0,  DHH), s1);
    s1 = wmma_bf16(qa1, load_b_frag(krow + (size_t)(jj0 + 16) * DHH + 32, DHH), s1);

    // ---- bias + mask + online softmax (row stats across 16-lane groups)
#pragma unroll
    for (int r = 0; r < 8; ++r) {
      const int i  = m0 + r + 8 * hi;       // query position
      const int jA = jj0 + col;
      const int jB = jj0 + 16 + col;
      float sa = (jA <= i + NNUL && jA < JJ)
                 ? (s0[r] * SCALE_F - slope * fabsf((float)(jA - (i + NNUL))))
                 : -INFINITY;
      float sb = (jB <= i + NNUL && jB < JJ)
                 ? (s1[r] * SCALE_F - slope * fabsf((float)(jB - (i + NNUL))))
                 : -INFINITY;
      float tmax = fmaxf(sa, sb);
#pragma unroll
      for (int mm = 8; mm >= 1; mm >>= 1) tmax = fmaxf(tmax, __shfl_xor(tmax, mm, 32));
      const float nm   = fmaxf(rmax[r], tmax);        // finite from block 0 onward
      const float corr = __expf(rmax[r] - nm);
      const float pa = __expf(sa - nm);
      const float pb = __expf(sb - nm);
      float psum = pa + pb;
#pragma unroll
      for (int mm = 8; mm >= 1; mm >>= 1) psum += __shfl_xor(psum, mm, 32);
      rsum[r] = rsum[r] * corr + psum;
      rmax[r] = nm;
      o0[r] *= corr; o1[r] *= corr; o2[r] *= corr; o3[r] *= corr;
      // stash P into wave-private LDS in row-major 16x32 (C-layout -> A-layout)
      pw[(r + 8 * hi) * 32 + col]      = f2bf(pa);
      pw[(r + 8 * hi) * 32 + 16 + col] = f2bf(pb);
    }

    // ---- O += P * V   (A = P 16x32 from LDS, B = V^T rows are vt rows)
    const v16bf pfrag = load_a_frag(pw, 32);
    o0 = wmma_bf16(pfrag, load_b_frag(vrow + (size_t)0  * JP + jj0, JP), o0);
    o1 = wmma_bf16(pfrag, load_b_frag(vrow + (size_t)16 * JP + jj0, JP), o1);
    o2 = wmma_bf16(pfrag, load_b_frag(vrow + (size_t)32 * JP + jj0, JP), o2);
    o3 = wmma_bf16(pfrag, load_b_frag(vrow + (size_t)48 * JP + jj0, JP), o3);
  }

  // ---- epilogue: O / l  -> aob[b][n][h*64 + d] (bf16, feeds the Wout GEMM)
#pragma unroll
  for (int r = 0; r < 8; ++r) {
    const float rinv = 1.0f / rsum[r];
    const size_t nrow = (size_t)b * NSEQ + (m0 + r + 8 * hi);
    bf16* dst = aob + nrow * INNER + h * DHH;
    dst[ 0 + col] = f2bf(o0[r] * rinv);
    dst[16 + col] = f2bf(o1[r] * rinv);
    dst[32 + col] = f2bf(o2[r] * rinv);
    dst[48 + col] = f2bf(o3[r] * rinv);
  }
}

// ---------------- host launcher ----------------
extern "C" void kernel_launch(void* const* d_in, const int* in_sizes, int n_in,
                              void* d_out, int out_size, void* d_ws, size_t ws_size,
                              hipStream_t stream) {
  (void)in_sizes; (void)n_in; (void)out_size; (void)ws_size;
  const float* x       = (const float*)d_in[0];
  const float* norm_g  = (const float*)d_in[1];
  const float* norm_b  = (const float*)d_in[2];
  const float* Wq      = (const float*)d_in[3];
  const float* Wkv     = (const float*)d_in[4];
  const float* q_scale = (const float*)d_in[5];
  const float* k_scale = (const float*)d_in[6];
  const float* null_kv = (const float*)d_in[7];
  const float* Wout    = (const float*)d_in[8];
  float* out = (float*)d_out;

  char* ws = (char*)d_ws;
  size_t off = 0;
  auto take = [&](size_t bytes) {
    char* p = ws + off;
    off += (bytes + 255) & ~(size_t)255;
    return p;
  };
  bf16*  xb    = (bf16*) take((size_t)BB * NSEQ * DIMM * 2);
  bf16*  xnb   = (bf16*) take((size_t)BB * NSEQ * DIMM * 2);
  bf16*  WqT   = (bf16*) take((size_t)INNER * DIMM * 2);
  bf16*  WkvT  = (bf16*) take((size_t)2 * INNER * DIMM * 2);
  bf16*  WoutT = (bf16*) take((size_t)DIMM * INNER * 2);
  float* qf    = (float*)take((size_t)BB * NSEQ * INNER * 4);
  float* kvf   = (float*)take((size_t)BB * NSEQ * 2 * INNER * 4);
  bf16*  qb    = (bf16*) take((size_t)BB * HH * NSEQ * DHH * 2);
  bf16*  kb    = (bf16*) take((size_t)BB * HH * JP * DHH * 2);
  bf16*  vt    = (bf16*) take((size_t)BB * HH * DHH * JP * 2);
  bf16*  aob   = (bf16*) take((size_t)BB * NSEQ * INNER * 2);

  const int M = BB * NSEQ;  // 4096

  // 1) LayerNorm + bf16 casts of x / xn
  ln_prep_kernel<<<M, 256, 0, stream>>>(x, norm_g, norm_b, xb, xnb);

  // 2) weight cast + transpose to [N][K] bf16
  cvt_transpose_kernel<<<4096, 256, 0, stream>>>(Wq,   WqT,   DIMM, INNER);
  cvt_transpose_kernel<<<8192, 256, 0, stream>>>(Wkv,  WkvT,  DIMM, 2 * INNER);
  cvt_transpose_kernel<<<4096, 256, 0, stream>>>(Wout, WoutT, INNER, DIMM);

  // 3) projections: q = xn @ Wq ; kv = x @ Wkv   (bf16 WMMA, f32 accum)
  {
    dim3 g(INNER / 256, M / 32);
    gemm_bf16_kernel<<<g, 256, 0, stream>>>(xnb, WqT, qf, M, INNER, DIMM);
  }
  {
    dim3 g(2 * INNER / 256, M / 32);
    gemm_bf16_kernel<<<g, 256, 0, stream>>>(xb, WkvT, kvf, M, 2 * INNER, DIMM);
  }

  // 4) l2norm + scale + null-kv splice, V transposed for B-fragment loads
  norm_q_kernel<<<(BB * NSEQ * HH) / 8, 256, 0, stream>>>(qf, q_scale, qb);
  {
    int rows = BB * HH * JJ;
    int blocks = (rows + 7) / 8;
    norm_k_kernel<<<blocks, 256, 0, stream>>>(kvf, null_kv, k_scale, kb);
    norm_v_kernel<<<blocks, 256, 0, stream>>>(kvf, null_kv, vt);
  }
  pad_v_kernel<<<240, 256, 0, stream>>>(vt);

  // 5) flash attention
  {
    dim3 g(NSEQ / 64, HH, BB);
    attn_kernel<<<g, 128, 0, stream>>>(qb, kb, vt, aob);
  }

  // 6) output projection: out = attn_out @ Wout  (fp32 result)
  {
    dim3 g(DIMM / 256, M / 32);
    gemm_bf16_kernel<<<g, 256, 0, stream>>>(aob, WoutT, out, M, DIMM, INNER);
  }
}
